// CSAttention_31971736551701
// MI455X (gfx1250) — matrix operand order repaired and checked by reading, not compiled, not run on hardware
//
#include <hip/hip_runtime.h>

// Problem dims
#define B_   128
#define S_   64
#define T_   50
#define F_   100
#define DA_  350
#define R_   30
#define HFC_ 100
#define H_   100

// WMMA-padded dims
#define TP   64
#define FP   128
#define DAP  352
#define RP   32

typedef __attribute__((ext_vector_type(16))) __bf16 v16bf;
typedef __attribute__((ext_vector_type(8)))  float  v8f;

// 16-bit A fragment (16x32, MxK) from row-major [m][k] LDS (ld in elements).
// lane L, halfword j -> k = (L/16)*16 + j, row m = L&15.
__device__ __forceinline__ v16bf ldsA(const __bf16* p0, int ld, int lane) {
  const int m  = lane & 15;
  const int kh = (lane >> 4) * 16;
  const __bf16* p = p0 + m * ld;
  v16bf a;
#pragma unroll
  for (int j = 0; j < 16; ++j) a[j] = p[kh + j];
  return a;
}

// 16-bit B fragment (32x16, KxN) from row-major [n][k] storage (i.e. the
// matrix is stored transposed: B(k,n) = mat[n][k]).
// lane L, halfword j -> k = (L/16)*16 + j, col n = L&15  => contiguous 32B.
__device__ __forceinline__ v16bf ldsB(const __bf16* p0, int ld, int lane) {
  const int n  = lane & 15;
  const int kb = (lane >> 4) * 16;
  const __bf16* p = p0 + n * ld + kb;
  v16bf b;
#pragma unroll
  for (int j = 0; j < 16; ++j) b[j] = p[j];
  return b;
}

// Branch-free tanh: clamp then (e^2x - 1)/(e^2x + 1).  Straight-line
// v_exp_f32 + v_rcp_f32, no divergent libm range-fix branches.
__device__ __forceinline__ float fast_tanh(float x) {
  float cx = fminf(fmaxf(x, -15.f), 15.f);
  float e  = __expf(2.f * cx);
  return __fdividef(e - 1.f, e + 1.f);
}

__device__ __forceinline__ float fast_sigmoid(float x) {
  return __fdividef(1.f, 1.f + __expf(-x));
}

// LDS layout offsets (bytes)
#define OFF_W    0                           // [DAP][FP] bf16  = 90112
#define OFF_U    (OFF_W   + DAP*FP*2)        // [RP][DAP] bf16  = 22528
#define OFF_XS   (OFF_U   + RP*DAP*2)        // [TP][FP]  bf16  = 16384
#define OFF_UIT  (OFF_XS  + TP*FP*2)         // [TP][DAP] bf16  = 45056
#define OFF_AIT  (OFF_UIT + TP*DAP*2)        // [TP][RP]  f32   = 8192
#define OFF_A    (OFF_AIT + TP*RP*4)         // [R_][52]  f32   = 6240
#define OFF_M    (OFF_A   + R_*52*4)         // [3008]    f32   = 12032
#define OFF_FCP  (OFF_M   + 3008*4)          // [2][112]  f32   = 896
#define OFF_FC   (OFF_FCP + 2*112*4)         // [112]     f32
#define OFF_H    (OFF_FC  + 112*4)           // [112]     f32
#define OFF_Z    (OFF_H   + 112*4)           // [112]     f32
#define OFF_R    (OFF_Z   + 112*4)           // [112]     f32
#define OFF_HP   (OFF_R   + 112*4)           // [112]     f32
#define OFF_CTX  (OFF_HP  + 112*4)           // [DAP]     f32   = 1408
#define SMEM_BYTES (OFF_CTX + DAP*4)         // ~205 KB < 320 KB WGP LDS

__global__ void __launch_bounds__(256)
csatt_scan_kernel(const float* __restrict__ x,   const float* __restrict__ W,
                  const float* __restrict__ Wc,  const float* __restrict__ bvec,
                  const float* __restrict__ u,   const float* __restrict__ Wfc,
                  const float* __restrict__ bfc,
                  const float* __restrict__ Wz,  const float* __restrict__ Wr,
                  const float* __restrict__ Wh,
                  const float* __restrict__ Uz,  const float* __restrict__ Ur,
                  const float* __restrict__ Uh,
                  const float* __restrict__ bz,  const float* __restrict__ br,
                  const float* __restrict__ bh,
                  float* __restrict__ out)
{
  extern __shared__ char smem[];
  __bf16* W_l   = (__bf16*)(smem + OFF_W);
  __bf16* u_l   = (__bf16*)(smem + OFF_U);
  __bf16* xs_l  = (__bf16*)(smem + OFF_XS);
  __bf16* uit_l = (__bf16*)(smem + OFF_UIT);
  float*  ait_l = (float*)(smem + OFF_AIT);
  float*  A_l   = (float*)(smem + OFF_A);
  float*  M_l   = (float*)(smem + OFF_M);
  float*  fcp_l = (float*)(smem + OFF_FCP);
  float*  fc_l  = (float*)(smem + OFF_FC);
  float*  h_l   = (float*)(smem + OFF_H);
  float*  z_l   = (float*)(smem + OFF_Z);
  float*  r_l   = (float*)(smem + OFF_R);
  float*  hp_l  = (float*)(smem + OFF_HP);
  float*  ctx_l = (float*)(smem + OFF_CTX);

  const int tid  = threadIdx.x;
  const int lane = tid & 31;
  const int wave = tid >> 5;
  const int b    = blockIdx.x;    // one batch row per workgroup

  // ---- one-time init: zero padded LDS, stage W and u as bf16 ----
  for (int i = tid; i < DAP * FP; i += 256) W_l[i] = (__bf16)0.f;
  for (int i = tid; i < RP * DAP; i += 256) u_l[i] = (__bf16)0.f;
  for (int i = tid; i < TP * FP;  i += 256) xs_l[i] = (__bf16)0.f;
  if (tid < 112) h_l[tid] = 0.f;
  __syncthreads();
  for (int i = tid; i < DA_ * F_; i += 256) {
    int d = i / F_, f = i % F_;
    W_l[d * FP + f] = (__bf16)W[i];
  }
  for (int i = tid; i < R_ * DA_; i += 256) {
    int r = i / DA_, d = i % DA_;
    u_l[r * DAP + d] = (__bf16)u[i];
  }
  __syncthreads();

  const float* xb = x + (size_t)b * S_ * T_ * F_;

  for (int s = 0; s < S_; ++s) {
    const float* xsrc = xb + (size_t)s * T_ * F_;

    // ---- stage x_s (bf16) + compute ctxb[d] = h @ Wc[d,:] + b[d] ----
    for (int i = tid; i < T_ * F_; i += 256) {
      int t = i / F_, f = i % F_;
      xs_l[t * FP + f] = (__bf16)xsrc[i];
    }
    if (s + 1 < S_) {  // prefetch next sentence slab (global_prefetch_b8)
      const char* nx = (const char*)(xsrc + T_ * F_);
      for (int i = tid * 128; i < T_ * F_ * 4; i += 256 * 128)
        __builtin_prefetch(nx + i, 0, 3);
    }
    for (int d = tid; d < DAP; d += 256) {
      float acc = 0.f;
      if (d < DA_) {
        const float* wr = Wc + (size_t)d * H_;
        for (int k = 0; k < H_; ++k) acc += h_l[k] * wr[k];
        acc += bvec[d];
      }
      ctx_l[d] = acc;
    }
    __syncthreads();

    // ---- uit = tanh(xs @ W^T + ctxb) (bf16 WMMA) ----
    // Each wave pins one t-tile (2 waves per t-tile) and hoists all four
    // K-step A fragments into registers; loops 11 n-tiles loading only B.
    {
      const int ti = wave & 3;
      const int n0 = wave >> 2;
      const __bf16* aBase = xs_l + (ti * 16) * FP;
      v16bf a0 = ldsA(aBase +  0, FP, lane);
      v16bf a1 = ldsA(aBase + 32, FP, lane);
      v16bf a2 = ldsA(aBase + 64, FP, lane);
      v16bf a3 = ldsA(aBase + 96, FP, lane);
      const int row0 = ti * 16 + (lane >> 4) * 8;
      const int nlo  = lane & 15;
      for (int ni = n0; ni < 22; ni += 2) {
        const __bf16* bBase = W_l + (ni * 16) * FP;
        v8f acc = {};
        acc = __builtin_amdgcn_wmma_f32_16x16x32_bf16(
            false, a0, false, ldsB(bBase +  0, FP, lane), (short)0, acc, false, false);
        acc = __builtin_amdgcn_wmma_f32_16x16x32_bf16(
            false, a1, false, ldsB(bBase + 32, FP, lane), (short)0, acc, false, false);
        acc = __builtin_amdgcn_wmma_f32_16x16x32_bf16(
            false, a2, false, ldsB(bBase + 64, FP, lane), (short)0, acc, false, false);
        acc = __builtin_amdgcn_wmma_f32_16x16x32_bf16(
            false, a3, false, ldsB(bBase + 96, FP, lane), (short)0, acc, false, false);
        const int col = ni * 16 + nlo;
        const float cb = ctx_l[col];
#pragma unroll
        for (int v = 0; v < 8; ++v)
          uit_l[(row0 + v) * DAP + col] = (__bf16)fast_tanh(acc[v] + cb);
      }
    }
    __syncthreads();

    // ---- ait = uit @ u^T : 4 x 2 tiles, K=352 (bf16 WMMA), one per wave ----
    {
      int ti = wave & 3, ri = wave >> 2;
      v8f acc = {};
      const __bf16* aBase = uit_l + (ti * 16) * DAP;
      const __bf16* bBase = u_l + (ri * 16) * DAP;
      for (int kb = 0; kb < DAP; kb += 32) {
        v16bf av = ldsA(aBase + kb, DAP, lane);
        v16bf bv = ldsB(bBase + kb, DAP, lane);
        acc = __builtin_amdgcn_wmma_f32_16x16x32_bf16(
            false, av, false, bv, (short)0, acc, false, false);
      }
      int row0 = ti * 16 + (lane >> 4) * 8;
      int col  = ri * 16 + (lane & 15);
#pragma unroll
      for (int v = 0; v < 8; ++v)
        ait_l[(row0 + v) * RP + col] = acc[v];
    }
    __syncthreads();

    // ---- softmax over the keras reshape (B,T,R)->(B,R,T): flat groups ----
    for (int g = wave; g < R_; g += 8) {
      int base = g * T_;
      int i0 = lane, i1 = lane + 32;
      float v0 = -1e30f, v1 = -1e30f;
      if (i0 < T_) { int n = base + i0; v0 = ait_l[(n / R_) * RP + (n % R_)]; }
      if (i1 < T_) { int n = base + i1; v1 = ait_l[(n / R_) * RP + (n % R_)]; }
      float mx = fmaxf(v0, v1);
      for (int off = 16; off > 0; off >>= 1)
        mx = fmaxf(mx, __shfl_xor(mx, off, 32));
      float e0 = (i0 < T_) ? __expf(v0 - mx) : 0.f;
      float e1 = (i1 < T_) ? __expf(v1 - mx) : 0.f;
      float sm = e0 + e1;
      for (int off = 16; off > 0; off >>= 1)
        sm += __shfl_xor(sm, off, 32);
      float inv = __fdividef(1.f, sm);
      if (i0 < T_) A_l[g * 52 + i0] = e0 * inv;
      if (i1 < T_) A_l[g * 52 + i1] = e1 * inv;
    }
    __syncthreads();

    // ---- M(r,f) = sum_t A(r,t) x(t,f)  (30x100, VALU) ----
    for (int o = tid; o < R_ * F_; o += 256) {
      int g = o / F_, f = o % F_;
      float acc = 0.f;
      for (int t = 0; t < T_; ++t)
        acc += A_l[g * 52 + t] * (float)xs_l[t * FP + f];
      M_l[o] = acc;
    }
    __syncthreads();

    // ---- fc = relu(Mflat @ Wfc + bfc) : split-K GEMV, coalesced on hf ----
    if (tid < 200) {
      int part = tid / 100, hf = tid % 100;
      int j0 = part * 1500, j1 = j0 + 1500;
      float acc = 0.f;
      for (int j = j0; j < j1; ++j)
        acc += M_l[j] * Wfc[(size_t)j * HFC_ + hf];
      fcp_l[part * 112 + hf] = acc;
    }
    __syncthreads();
    if (tid < HFC_)
      fc_l[tid] = fmaxf(fcp_l[tid] + fcp_l[112 + tid] + bfc[tid], 0.f);
    __syncthreads();

    // ---- GRU gates z, r ----
    if (tid < 200) {
      int which = tid / 100, n = tid % 100;
      const float* Wm = which ? Wr : Wz;
      const float* Um = which ? Ur : Uz;
      const float* bm = which ? br : bz;
      float acc = bm[n];
      for (int k = 0; k < HFC_; ++k) acc += fc_l[k] * Wm[k * H_ + n];
      for (int k = 0; k < H_;  ++k) acc += h_l[k] * Um[k * H_ + n];
      (which ? r_l : z_l)[n] = fast_sigmoid(acc);
    }
    __syncthreads();
    // ---- candidate hp ----
    if (tid < H_) {
      int n = tid;
      float acc = bh[n];
      for (int k = 0; k < HFC_; ++k) acc += fc_l[k] * Wh[k * H_ + n];
      for (int k = 0; k < H_;  ++k) acc += (r_l[k] * h_l[k]) * Uh[k * H_ + n];
      hp_l[n] = fast_tanh(acc);
    }
    __syncthreads();
    // ---- h_new = (1-z)h + z*hp ; emit output ----
    if (tid < H_) {
      int n = tid;
      float z  = z_l[n];
      float hn = (1.f - z) * h_l[n] + z * hp_l[n];
      h_l[n] = hn;
      out[((size_t)b * S_ + s) * H_ + n] = hn;
    }
    __syncthreads();
  }
}

extern "C" void kernel_launch(void* const* d_in, const int* in_sizes, int n_in,
                              void* d_out, int out_size, void* d_ws, size_t ws_size,
                              hipStream_t stream) {
  (void)in_sizes; (void)n_in; (void)out_size; (void)d_ws; (void)ws_size;
  const float* x    = (const float*)d_in[0];
  const float* W    = (const float*)d_in[1];
  const float* Wc   = (const float*)d_in[2];
  const float* bvec = (const float*)d_in[3];
  const float* u    = (const float*)d_in[4];
  const float* Wfc  = (const float*)d_in[5];
  const float* bfc  = (const float*)d_in[6];
  const float* Wz   = (const float*)d_in[7];
  const float* Wr   = (const float*)d_in[8];
  const float* Wh   = (const float*)d_in[9];
  const float* Uz   = (const float*)d_in[10];
  const float* Ur   = (const float*)d_in[11];
  const float* Uh   = (const float*)d_in[12];
  const float* bz   = (const float*)d_in[13];
  const float* br   = (const float*)d_in[14];
  const float* bh   = (const float*)d_in[15];
  float* out = (float*)d_out;

  csatt_scan_kernel<<<B_, 256, SMEM_BYTES, stream>>>(
      x, W, Wc, bvec, u, Wfc, bfc, Wz, Wr, Wh, Uz, Ur, Uh, bz, br, bh, out);
}
